// FGEncoder_3813930959340
// MI455X (gfx1250) — matrix operand order repaired and checked by reading, not compiled, not run on hardware
//
#include <hip/hip_runtime.h>
#include <hip/hip_bf16.h>

typedef _Float16 half_t;
typedef __attribute__((ext_vector_type(16))) _Float16 v16h;
typedef __attribute__((ext_vector_type(8)))  _Float16 v8h;
typedef __attribute__((ext_vector_type(4)))  _Float16 v4h;
typedef __attribute__((ext_vector_type(8)))  float    v8f;
typedef __attribute__((ext_vector_type(4)))  float    v4f;

#define TMAX 512
#define DIN  512
#define DHID 256
#define DOUT 3

// ---------------------------------------------------------------------------
// Kernel 1: per-batch duration scan.  One block per batch, 512 threads.
// d = max(floor(ds * L/Lmax), 1) when ds>0 else 0; exclusive cumsum -> starts.
// ---------------------------------------------------------------------------
__global__ void fg_scan_kernel(const int* __restrict__ ds,
                               const int* __restrict__ lmax_p,
                               int L,
                               int* __restrict__ starts,
                               int* __restrict__ lens)
{
    __shared__ int sh[TMAX];
    const int b = blockIdx.x;
    const int t = threadIdx.x;

    const int   dsv  = ds[b * TMAX + t];
    const float mult = (float)L / (float)lmax_p[0];
    int d = (int)floorf((float)dsv * mult);
    if (d < 1) d = 1;
    const int deff = (dsv > 0) ? d : 0;

    sh[t] = deff;
    __syncthreads();
    // Hillis-Steele inclusive scan over 512 entries
    for (int off = 1; off < TMAX; off <<= 1) {
        int v = (t >= off) ? sh[t - off] : 0;
        __syncthreads();
        sh[t] += v;
        __syncthreads();
    }
    const int incl  = sh[t];
    int start = incl - deff;
    int s_cl = start;        if (s_cl > L) s_cl = L; if (s_cl < 0) s_cl = 0;
    int e_cl = start + deff; if (e_cl > L) e_cl = L; if (e_cl < 0) e_cl = 0;
    int len  = (dsv > 0) ? (e_cl - s_cl) : 0;   // len<=0  =>  output row is zero
    starts[b * TMAX + t] = s_cl;
    lens  [b * TMAX + t] = len;
}

// ---------------------------------------------------------------------------
// Kernel 2: pre-swizzle W1 (f32, K x N row-major, 512x256) into WMMA B-fragment
// order, converted to f16.  Layout: Bp[((nt*16 + ks)*32 + lane)*16 + j]
//   lane 0-15  : column n = nt*16+lane, K = ks*32 + {0..7, 16..23}   (j=0..15)
//   lane 16-31 : column n = nt*16+(lane-16), K = ks*32 + {8..15, 24..31}
// ---------------------------------------------------------------------------
__global__ void fg_pack_w1_kernel(const float* __restrict__ W1,
                                  half_t* __restrict__ Bp)
{
    const int idx = blockIdx.x * blockDim.x + threadIdx.x;   // 0 .. 131071
    if (idx >= 16 * 16 * 32 * 16) return;
    const int j    =  idx        & 15;
    const int lane = (idx >> 4)  & 31;
    const int ks   = (idx >> 9)  & 15;
    const int nt   =  idx >> 13;
    const int k = ks * 32 + ((lane >= 16) ? 8 : 0) + (j & 7) + ((j & 8) ? 16 : 0);
    const int n = nt * 16 + (lane & 15);
    Bp[idx] = (half_t)W1[k * DHID + n];
}

// ---------------------------------------------------------------------------
// Kernel 3: segment mean -> f16 activation row.  One block per (b,t) row,
// 128 threads x 4 channels (float4 loads, v4h store).
// ---------------------------------------------------------------------------
__global__ void fg_align_kernel(const float* __restrict__ hs,
                                const int* __restrict__ starts,
                                const int* __restrict__ lens,
                                int L,
                                half_t* __restrict__ A)
{
    const int row = blockIdx.x;            // b*TMAX + t
    const int b   = row / TMAX;
    const int c0  = threadIdx.x * 4;       // 128 * 4 = 512 channels
    const int start = starts[row];
    const int len   = lens[row];

    v4f acc = {0.f, 0.f, 0.f, 0.f};
    const float* base = hs + ((size_t)b * (size_t)L + (size_t)start) * DIN + c0;
    for (int f = 0; f < len; ++f) {
        v4f x = *(const v4f*)(base + (size_t)f * DIN);
        acc += x;
    }
    const float inv = (len > 0) ? (1.0f / (float)len) : 0.0f;
    v4h o;
    o[0] = (half_t)(acc[0] * inv);
    o[1] = (half_t)(acc[1] * inv);
    o[2] = (half_t)(acc[2] * inv);
    o[3] = (half_t)(acc[3] * inv);
    *(v4h*)(A + (size_t)row * DIN + c0) = o;
}

// ---------------------------------------------------------------------------
// Kernel 4: fused  relu(A @ W1 + b1) @ W2 + b2, relu  via WMMA f16->f32.
// Block = 128 threads = 4 wave32; handles 16 rows.  Wave w owns N-columns
// [w*64, w*64+64) of the hidden tile (4 WMMA tiles).  K-loop fully unrolled
// (16 x K32 x 4 tiles = 64 WMMAs/wave) so accumulators stay in fixed VGPRs.
// Hidden tile never leaves registers; GEMM2 (N=3) reduced via LDS f32 atomics.
// ---------------------------------------------------------------------------
__global__ void __launch_bounds__(128)
fg_mlp_kernel(const half_t* __restrict__ A,   // rows x 512
              const half_t* __restrict__ Bp,  // packed W1 frags
              const float*  __restrict__ b1,
              const float*  __restrict__ W2,  // 256 x 3
              const float*  __restrict__ b2,
              float* __restrict__ out)        // rows x 3
{
    const int lane = threadIdx.x & 31;
    const int wave = threadIdx.x >> 5;         // 0..3
    const int r0   = blockIdx.x * 16;
    const int mrow = lane & 15;
    const bool hi  = lane >= 16;

    __shared__ float acc_out[16 * DOUT];
    if (threadIdx.x < 16 * DOUT) acc_out[threadIdx.x] = 0.0f;
    __syncthreads();

    v8f c[4] = {};

    // A-fragment base for this lane (ISA 16-bit 16x32 A layout)
    const half_t* arow = A + (size_t)(r0 + mrow) * DIN + (hi ? 8 : 0);
    const half_t* bbase = Bp + ((size_t)(wave * 4) * 16 * 32 + lane) * 16;

#pragma unroll
    for (int ks = 0; ks < 16; ++ks) {
        const half_t* ap = arow + ks * 32;
        v8h alo = *(const v8h*)(ap);        // K = k0+{0..7}   (hi: k0+{8..15})
        v8h ahi = *(const v8h*)(ap + 16);   // K = k0+{16..23} (hi: k0+{24..31})
        v16h a;
#pragma unroll
        for (int j = 0; j < 8; ++j) { a[j] = alo[j]; a[8 + j] = ahi[j]; }

#pragma unroll
        for (int i = 0; i < 4; ++i) {
            const v16h b = *(const v16h*)(bbase + ((size_t)(i * 16 + ks) * 32) * 16);
            c[i] = __builtin_amdgcn_wmma_f32_16x16x32_f16(
                       false, a, false, b, (short)0, c[i], false, false);
        }
    }

    // bias1 + relu + GEMM2 partials (each lane holds column n, rows m..m+7)
    float part[8][DOUT];
#pragma unroll
    for (int v = 0; v < 8; ++v)
#pragma unroll
        for (int o = 0; o < DOUT; ++o) part[v][o] = 0.0f;

#pragma unroll
    for (int i = 0; i < 4; ++i) {
        const int n = wave * 64 + i * 16 + (lane & 15);
        const float bias = b1[n];
        const float w0 = W2[n * DOUT + 0];
        const float w1 = W2[n * DOUT + 1];
        const float w2 = W2[n * DOUT + 2];
#pragma unroll
        for (int v = 0; v < 8; ++v) {
            float h = c[i][v] + bias;
            h = fmaxf(h, 0.0f);
            part[v][0] += h * w0;
            part[v][1] += h * w1;
            part[v][2] += h * w2;
        }
    }
#pragma unroll
    for (int v = 0; v < 8; ++v) {
        const int row = v + (hi ? 8 : 0);
        atomicAdd(&acc_out[row * DOUT + 0], part[v][0]);
        atomicAdd(&acc_out[row * DOUT + 1], part[v][1]);
        atomicAdd(&acc_out[row * DOUT + 2], part[v][2]);
    }
    __syncthreads();

    if (threadIdx.x < 16 * DOUT) {
        const int row = threadIdx.x / DOUT;
        const int o   = threadIdx.x % DOUT;
        float r = fmaxf(acc_out[threadIdx.x] + b2[o], 0.0f);
        out[(size_t)(r0 + row) * DOUT + o] = r;
    }
}

// ---------------------------------------------------------------------------
extern "C" void kernel_launch(void* const* d_in, const int* in_sizes, int n_in,
                              void* d_out, int out_size, void* d_ws, size_t ws_size,
                              hipStream_t stream) {
    const float* hs   = (const float*)d_in[0];
    const int*   ds   = (const int*)  d_in[1];
    const float* W1   = (const float*)d_in[2];
    const float* b1   = (const float*)d_in[3];
    const float* W2   = (const float*)d_in[4];
    const float* b2   = (const float*)d_in[5];
    const int*   lmax = (const int*)  d_in[6];

    const int B    = in_sizes[1] / TMAX;            // 16
    const int L    = in_sizes[0] / (B * DIN);       // 4096
    const int rows = B * TMAX;                      // 8192

    char* ws = (char*)d_ws;
    half_t* A  = (half_t*)ws;                                    // rows*512 f16
    size_t offA = (size_t)rows * DIN * sizeof(half_t);
    half_t* Bp = (half_t*)(ws + offA);                           // 256 KB packed W1
    size_t offB = offA + (size_t)16 * 16 * 32 * 16 * sizeof(half_t);
    int* starts = (int*)(ws + offB);
    int* lens   = starts + rows;

    fg_scan_kernel   <<<B, TMAX, 0, stream>>>(ds, lmax, L, starts, lens);
    fg_pack_w1_kernel<<<(16 * 16 * 32 * 16) / 256, 256, 0, stream>>>(W1, Bp);
    fg_align_kernel  <<<rows, 128, 0, stream>>>(hs, starts, lens, L, A);
    fg_mlp_kernel    <<<rows / 16, 128, 0, stream>>>(A, Bp, b1, W2, b2, (float*)d_out);
}
